// NSAEncoder_32908039422416
// MI455X (gfx1250) — matrix operand order repaired and checked by reading, not compiled, not run on hardware
//
#include <hip/hip_runtime.h>
#include <hip/hip_bf16.h>
#include <math.h>

// ---------------------------------------------------------------------------
// NSAEncoder on gfx1250 (MI455X): wave32, v_wmma_f32_16x16x32_f16 everywhere,
// all WMMA operands built from 16-byte ds_load_b128-friendly LDS layouts.
// ---------------------------------------------------------------------------

typedef __attribute__((ext_vector_type(16))) _Float16 v16h;
typedef __attribute__((ext_vector_type(8)))  _Float16 v8h;
typedef __attribute__((ext_vector_type(8)))  float    v8f;

#define NGRAPH   4
#define NPG      2048
#define NNODES   (NGRAPH * NPG)
#define IN_DIM   64
#define HID      512
#define HEADS    8
#define HD       64
#define LAYERS   4
#define LN_EPS   1e-5f
#define ADJ_WPR  (NPG / 32)     // 64 words of adjacency bits per query row

static __device__ __forceinline__ v8f wmma_f16(v16h a, v16h b, v8f c) {
    return __builtin_amdgcn_wmma_f32_16x16x32_f16(false, a, false, b,
                                                  (short)0, c, false, false);
}
static __device__ __forceinline__ v16h cat8(v8h lo, v8h hi) {
    return __builtin_shufflevector(lo, hi, 0, 1, 2, 3, 4, 5, 6, 7,
                                           8, 9, 10, 11, 12, 13, 14, 15);
}
static __device__ __forceinline__ v8f v8f_zero() {
    return (v8f){0.f, 0.f, 0.f, 0.f, 0.f, 0.f, 0.f, 0.f};
}

// --------------------------------- utility ---------------------------------

__global__ void zero_u32(unsigned* __restrict__ p, int n) {
    int i = blockIdx.x * blockDim.x + threadIdx.x;
    if (i < n) p[i] = 0u;
}

__global__ void build_adj(const int* __restrict__ ei, unsigned* __restrict__ adj, int E) {
    int e = blockIdx.x * blockDim.x + threadIdx.x;
    if (e >= E) return;
    int src = ei[e];
    int dst = ei[E + e];
    int b  = src / NPG;               // batch[src]
    int si = src - b * NPG;
    int di = dst - b * NPG;           // dst constructed within same graph
    atomicOr(&adj[((size_t)b * NPG + si) * ADJ_WPR + (di >> 5)], 1u << (di & 31));
}

__global__ void f32_to_f16(const float* __restrict__ in, _Float16* __restrict__ out, size_t n) {
    size_t i = (size_t)blockIdx.x * blockDim.x + threadIdx.x;
    size_t stride = (size_t)gridDim.x * blockDim.x;
    for (; i < n; i += stride) out[i] = (_Float16)in[i];
}

// ------------------------------ layer norm ---------------------------------
// one row (HID=512) per 128-thread block; output f16

__global__ __launch_bounds__(128)
void layernorm_f16(const float* __restrict__ h, const float* __restrict__ g,
                   const float* __restrict__ bt, _Float16* __restrict__ out) {
    int row = blockIdx.x;
    int tid = threadIdx.x;
    int lane = tid & 31, wave = tid >> 5;
    const float* hr = h + (size_t)row * HID;

    float v[4];
    float s = 0.f;
#pragma unroll
    for (int k = 0; k < 4; ++k) { v[k] = hr[tid + k * 128]; s += v[k]; }

    __shared__ float red[4];
#pragma unroll
    for (int off = 16; off; off >>= 1) s += __shfl_xor(s, off, 32);
    if (lane == 0) red[wave] = s;
    __syncthreads();
    float mu = (red[0] + red[1] + red[2] + red[3]) * (1.f / HID);
    __syncthreads();

    float q = 0.f;
#pragma unroll
    for (int k = 0; k < 4; ++k) { float d = v[k] - mu; q += d * d; }
#pragma unroll
    for (int off = 16; off; off >>= 1) q += __shfl_xor(q, off, 32);
    if (lane == 0) red[wave] = q;
    __syncthreads();
    float var = (red[0] + red[1] + red[2] + red[3]) * (1.f / HID);
    float w = rsqrtf(var + LN_EPS);

    _Float16* orow = out + (size_t)row * HID;
#pragma unroll
    for (int k = 0; k < 4; ++k) {
        int c = tid + k * 128;
        orow[c] = (_Float16)((v[k] - mu) * w * g[c] + bt[c]);
    }
}

// ------------------------------- WMMA GEMM ---------------------------------
// C[M,N] = A[M,K](f16,rm) @ B[K,N](f16,rm) + bias
// MODE 0: Cf = v            (projection)
// MODE 1: Ch = f16(gelu(v)) (MLP up, exact erf GELU)
// MODE 2: Cf += v           (MLP down + residual)
// Block 256 = 8 waves; tile 128x128x32; wave owns 32x64 (2x4 subtiles).
// B is staged *transposed* so every fragment is two 16B contiguous LDS loads.
// Global loads for k-step t+1 are issued while computing k-step t.

template <int MODE>
__global__ __launch_bounds__(256)
void gemm_wmma(const _Float16* __restrict__ A, const _Float16* __restrict__ Bm,
               const float* __restrict__ bias, float* __restrict__ Cf,
               _Float16* __restrict__ Ch, int M, int N, int K) {
    __shared__ _Float16 As[128][40];   // [m][k], 80B row stride (16B aligned)
    __shared__ _Float16 Bt[128][40];   // [n][k], transposed B tile

    const int tid  = threadIdx.x;
    const int lane = tid & 31, wave = tid >> 5;
    const int r = lane & 15, half = lane >> 4, kh = half * 16;
    const int wm = wave >> 1, wn = wave & 1;
    const int m0 = blockIdx.y * 128, n0 = blockIdx.x * 128;

    // staging coordinates (2 x v8h each for A and B per thread)
    int aR[2], aC[2], bR[2], bC[2];
#pragma unroll
    for (int v = 0; v < 2; ++v) {
        int idx = tid + v * 256;                  // 0..511
        aR[v] = idx >> 2;  aC[v] = (idx & 3) * 8;   // A: 128 rows x 32 k
        bR[v] = idx >> 4;  bC[v] = (idx & 15) * 8;  // B: 32 k x 128 n
    }

    v8h aReg[2], bReg[2];
#pragma unroll
    for (int v = 0; v < 2; ++v) {
        aReg[v] = *(const v8h*)(A  + (size_t)(m0 + aR[v]) * K + aC[v]);
        bReg[v] = *(const v8h*)(Bm + (size_t)bR[v] * N + n0 + bC[v]);
    }

    v8f acc[2][4];
#pragma unroll
    for (int i = 0; i < 2; ++i)
#pragma unroll
        for (int j = 0; j < 4; ++j) acc[i][j] = v8f_zero();

    for (int k0 = 0; k0 < K; k0 += 32) {
        // commit staged registers to LDS (A vector, B transposed scatter)
#pragma unroll
        for (int v = 0; v < 2; ++v) {
            *(v8h*)&As[aR[v]][aC[v]] = aReg[v];
#pragma unroll
            for (int j = 0; j < 8; ++j) Bt[bC[v] + j][bR[v]] = bReg[v][j];
        }
        __syncthreads();

        // software pipeline: issue next tile's global loads now
        if (k0 + 32 < K) {
#pragma unroll
            for (int v = 0; v < 2; ++v) {
                aReg[v] = *(const v8h*)(A  + (size_t)(m0 + aR[v]) * K + (k0 + 32) + aC[v]);
                bReg[v] = *(const v8h*)(Bm + (size_t)(k0 + 32 + bR[v]) * N + n0 + bC[v]);
            }
        }

        v16h af[2], bf[4];
#pragma unroll
        for (int rs = 0; rs < 2; ++rs) {
            const _Float16* p = &As[wm * 32 + rs * 16 + r][0];
            af[rs] = cat8(*(const v8h*)(p + half * 8), *(const v8h*)(p + 16 + half * 8));
        }
#pragma unroll
        for (int cs = 0; cs < 4; ++cs) {
            const _Float16* p = &Bt[wn * 64 + cs * 16 + r][0];
            bf[cs] = cat8(*(const v8h*)(p + half * 8), *(const v8h*)(p + 16 + half * 8));
        }
#pragma unroll
        for (int rs = 0; rs < 2; ++rs)
#pragma unroll
            for (int cs = 0; cs < 4; ++cs)
                acc[rs][cs] = wmma_f16(af[rs], bf[cs], acc[rs][cs]);
        __syncthreads();
    }

#pragma unroll
    for (int rs = 0; rs < 2; ++rs)
#pragma unroll
        for (int cs = 0; cs < 4; ++cs)
#pragma unroll
            for (int e = 0; e < 8; ++e) {
                int row = m0 + wm * 32 + rs * 16 + e + 8 * half;
                int col = n0 + wn * 64 + cs * 16 + r;
                float v = acc[rs][cs][e] + bias[col];
                size_t gi = (size_t)row * N + col;
                if (MODE == 0) {
                    Cf[gi] = v;
                } else if (MODE == 1) {
                    float ge = 0.5f * v * (1.f + erff(v * 0.70710678118654752f));
                    Ch[gi] = (_Float16)ge;
                } else {
                    Cf[gi] += v;
                }
            }
}

// --------------------------- flash attention -------------------------------
// q = k = v = normed. grid: (NPG/64, NGRAPH*HEADS); block 128 = 4 waves,
// each wave owns 16 query rows. Streams 32-key blocks through LDS.
// kv   [key][dim] feeds score B-frags (contiguous dim runs),
// kvT  [dim][key] feeds PV   B-frags (contiguous key runs),
// pbuf per-wave P transpose; same-wave RAW fenced with s_wait_dscnt 0.

__global__ __launch_bounds__(128)
void attn_wmma(const _Float16* __restrict__ nrm, float* __restrict__ h,
               const unsigned* __restrict__ adj, int masked) {
    __shared__ _Float16 kv[32][72];        // 32 keys x 64 dims (144B stride)
    __shared__ _Float16 kvT[64][40];       // 64 dims x 32 keys (80B stride)
    __shared__ _Float16 pbuf[4][16][40];   // per-wave P tile (A-layout source)

    const int bh = blockIdx.y;
    const int b = bh / HEADS, head = bh % HEADS;
    const int tid = threadIdx.x;
    const int lane = tid & 31, wave = tid >> 5;
    const int r = lane & 15, half = lane >> 4, kh = half * 16;
    const int hc = head * HD;
    const size_t rowBase = (size_t)b * NPG;
    const int q0 = blockIdx.x * 64 + wave * 16;

    // Q fragments: 16 rows x 64 dims = 2 A-frags; 4 vector global loads
    const _Float16* qrow = nrm + (rowBase + q0 + r) * HID + hc;
    v16h qf[2];
#pragma unroll
    for (int fk = 0; fk < 2; ++fk)
        qf[fk] = cat8(*(const v8h*)(qrow + half * 8 + fk * 32),
                      *(const v8h*)(qrow + 16 + half * 8 + fk * 32));

    v8f o[4];
#pragma unroll
    for (int t = 0; t < 4; ++t) o[t] = v8f_zero();
    float m[8], l[8];
#pragma unroll
    for (int e = 0; e < 8; ++e) { m[e] = -1e30f; l[e] = 0.f; }

    // staging coordinates: 2 x v8h per thread covers 32x64 halves
    int sR[2], sC[2];
#pragma unroll
    for (int v = 0; v < 2; ++v) {
        int idx = tid + v * 128;               // 0..255
        sR[v] = idx >> 3; sC[v] = (idx & 7) * 8;
    }

    for (int kb = 0; kb < NPG / 32; ++kb) {
        __syncthreads();   // all waves done reading kv/kvT of previous block
#pragma unroll
        for (int v = 0; v < 2; ++v) {
            v8h d = *(const v8h*)(nrm + (rowBase + (size_t)kb * 32 + sR[v]) * HID + hc + sC[v]);
            *(v8h*)&kv[sR[v]][sC[v]] = d;
#pragma unroll
            for (int j = 0; j < 8; ++j) kvT[sC[v] + j][sR[v]] = d[j];
        }
        __syncthreads();

        // scores: s[ks] covers keys kb*32 + ks*16 .. +15
        v8f s[2];
#pragma unroll
        for (int ks = 0; ks < 2; ++ks) {
            v8f c = v8f_zero();
#pragma unroll
            for (int fk = 0; fk < 2; ++fk) {
                const _Float16* p = &kv[ks * 16 + r][fk * 32];
                c = wmma_f16(qf[fk], cat8(*(const v8h*)(p + half * 8), *(const v8h*)(p + 16 + half * 8)), c);
            }
            s[ks] = c;
        }

        // mask + scale; row max across 16-lane groups
        float mx[8];
#pragma unroll
        for (int e = 0; e < 8; ++e) {
            unsigned aw = 0xFFFFFFFFu;
            if (masked) {
                int ql = q0 + e + 8 * half;
                aw = adj[(rowBase + ql) * ADJ_WPR + kb];
            }
            float v0 = s[0][e] * 0.125f;           // 1/sqrt(64)
            float v1 = s[1][e] * 0.125f;
            if (!((aw >> r) & 1u))        v0 = -1e30f;
            if (!((aw >> (16 + r)) & 1u)) v1 = -1e30f;
            s[0][e] = v0; s[1][e] = v1;
            mx[e] = fmaxf(v0, v1);
        }
#pragma unroll
        for (int off = 1; off < 16; off <<= 1)
#pragma unroll
            for (int e = 0; e < 8; ++e) mx[e] = fmaxf(mx[e], __shfl_xor(mx[e], off, 32));

        float alpha[8], rsum[8];
#pragma unroll
        for (int e = 0; e < 8; ++e) {
            float mn = fmaxf(m[e], mx[e]);
            alpha[e] = __expf(m[e] - mn);
            m[e] = mn;
            float p0 = __expf(s[0][e] - mn);
            float p1 = __expf(s[1][e] - mn);
            s[0][e] = p0; s[1][e] = p1;
            rsum[e] = p0 + p1;
        }
#pragma unroll
        for (int off = 1; off < 16; off <<= 1)
#pragma unroll
            for (int e = 0; e < 8; ++e) rsum[e] += __shfl_xor(rsum[e], off, 32);
#pragma unroll
        for (int e = 0; e < 8; ++e) l[e] = l[e] * alpha[e] + rsum[e];
#pragma unroll
        for (int t = 0; t < 4; ++t)
#pragma unroll
            for (int e = 0; e < 8; ++e) o[t][e] *= alpha[e];

        // transpose P (C-layout) -> A-layout via per-wave LDS
#pragma unroll
        for (int e = 0; e < 8; ++e) {
            pbuf[wave][e + 8 * half][r]      = (_Float16)s[0][e];
            pbuf[wave][e + 8 * half][16 + r] = (_Float16)s[1][e];
        }
        // same-wave RAW on LDS: DS pipe is in-order per wave; fence compiler +
        // outstanding DS explicitly (cheaper than a block barrier)
        asm volatile("s_wait_dscnt 0" ::: "memory");

        const _Float16* pp = &pbuf[wave][r][0];
        v16h pf = cat8(*(const v8h*)(pp + half * 8), *(const v8h*)(pp + 16 + half * 8));
#pragma unroll
        for (int t = 0; t < 4; ++t) {
            const _Float16* vp = &kvT[t * 16 + r][0];
            o[t] = wmma_f16(pf, cat8(*(const v8h*)(vp + half * 8), *(const v8h*)(vp + 16 + half * 8)), o[t]);
        }
    }

    // normalize + residual add into h (this block exclusively owns the region)
#pragma unroll
    for (int e = 0; e < 8; ++e) {
        float inv = 1.f / l[e];
        int ql = q0 + e + 8 * half;
#pragma unroll
        for (int t = 0; t < 4; ++t) {
            size_t gi = (rowBase + ql) * HID + hc + t * 16 + r;
            h[gi] += o[t][e] * inv;
        }
    }
}

// ------------------------------- launcher ----------------------------------

extern "C" void kernel_launch(void* const* d_in, const int* in_sizes, int n_in,
                              void* d_out, int out_size, void* d_ws, size_t ws_size,
                              hipStream_t stream) {
    const float* x      = (const float*)d_in[0];
    const int*   ei     = (const int*)  d_in[1];
    const float* proj_W = (const float*)d_in[3];
    const float* proj_b = (const float*)d_in[4];
    const float* ln1_g  = (const float*)d_in[5];
    const float* ln1_b  = (const float*)d_in[6];
    const float* ln2_g  = (const float*)d_in[7];
    const float* ln2_b  = (const float*)d_in[8];
    const float* W1     = (const float*)d_in[9];
    const float* b1     = (const float*)d_in[10];
    const float* W2     = (const float*)d_in[11];
    const float* b2     = (const float*)d_in[12];

    float* h = (float*)d_out;               // [NNODES, HID] fp32, lives in d_out

    // workspace carve-up (all offsets 256B aligned)
    char* ws = (char*)d_ws;
    unsigned*  adj    = (unsigned*) (ws + 0);                     //  2 MB
    _Float16*  xh     = (_Float16*) (ws + 2097152);               //  1 MB
    _Float16*  pWh    = (_Float16*) (ws + 3145728);               // 64 KB
    _Float16*  W1h    = (_Float16*) (ws + 3211264);               //  8 MB
    _Float16*  W2h    = (_Float16*) (ws + 11599872);              //  8 MB
    _Float16*  nrm    = (_Float16*) (ws + 19988480);              //  8 MB
    _Float16*  hidden = (_Float16*) (ws + 28377088);              // 32 MB

    const int E = in_sizes[1] / 2;          // edges incl. self loops

    // adjacency bitmask
    zero_u32<<<(NGRAPH * NPG * ADJ_WPR + 255) / 256, 256, 0, stream>>>(
        adj, NGRAPH * NPG * ADJ_WPR);
    build_adj<<<(E + 255) / 256, 256, 0, stream>>>(ei, adj, E);

    // f16 copies of activations-in and weights
    f32_to_f16<<<1024, 256, 0, stream>>>(x, xh, (size_t)NNODES * IN_DIM);
    f32_to_f16<<<64, 256, 0, stream>>>(proj_W, pWh, (size_t)IN_DIM * HID);
    f32_to_f16<<<2048, 256, 0, stream>>>(W1, W1h, (size_t)LAYERS * HID * 4 * HID);
    f32_to_f16<<<2048, 256, 0, stream>>>(W2, W2h, (size_t)LAYERS * 4 * HID * HID);

    // h = x @ proj_W + proj_b
    gemm_wmma<0><<<dim3(HID / 128, NNODES / 128), 256, 0, stream>>>(
        xh, pWh, proj_b, h, nullptr, NNODES, HID, IN_DIM);

    const int pattern_masked[LAYERS] = {1, 0, 1, 0};   // 'M','S','M','S'
    for (int i = 0; i < LAYERS; ++i) {
        layernorm_f16<<<NNODES, 128, 0, stream>>>(h, ln1_g + i * HID, ln1_b + i * HID, nrm);
        attn_wmma<<<dim3(NPG / 64, NGRAPH * HEADS), 128, 0, stream>>>(
            nrm, h, adj, pattern_masked[i]);
        layernorm_f16<<<NNODES, 128, 0, stream>>>(h, ln2_g + i * HID, ln2_b + i * HID, nrm);
        gemm_wmma<1><<<dim3(4 * HID / 128, NNODES / 128), 256, 0, stream>>>(
            nrm, W1h + (size_t)i * HID * 4 * HID, b1 + (size_t)i * 4 * HID,
            nullptr, hidden, NNODES, 4 * HID, HID);
        gemm_wmma<2><<<dim3(HID / 128, NNODES / 128), 256, 0, stream>>>(
            hidden, W2h + (size_t)i * 4 * HID * HID, b2 + (size_t)i * HID,
            h, nullptr, NNODES, HID, 4 * HID);
    }
    (void)ws_size; (void)out_size; (void)n_in;
}
